// BiDirectionalAddBlock_59622736003619
// MI455X (gfx1250) — matrix operand order repaired and checked by reading, not compiled, not run on hardware
//
#include <hip/hip_runtime.h>
#include <hip/hip_bf16.h>
#include <math.h>

// ---------------- problem constants ----------------
#define BB     2
#define LL     4096
#define DM     512          // D_MODEL
#define DI     1024         // D_INNER
#define DS     16           // D_STATE
#define DTR    32           // DT_RANK
#define MR     (BB*LL)      // 8192 rows (tokens)
#define CH     64           // scan chunk length
#define NC     (LL/CH)      // chunks per sequence (64)

typedef __attribute__((ext_vector_type(16))) __bf16 v16bf;
typedef __attribute__((ext_vector_type(8)))  __bf16 v8bf;
typedef __attribute__((ext_vector_type(8)))  float  v8f;

// ---------------- layernorm -> bf16 (normal + flipped copy) ----------------
__global__ __launch_bounds__(256) void ln_kernel(const float* __restrict__ x,
    const float* __restrict__ w, const float* __restrict__ bsh,
    __bf16* __restrict__ hn, __bf16* __restrict__ hn_rev)
{
    int row = blockIdx.x;                 // b*L + l
    int bb = row / LL, l = row % LL;
    const float* xr = x + (size_t)row * DM;
    float s = 0.f, s2 = 0.f;
    for (int i = threadIdx.x; i < DM; i += 256) { float v = xr[i]; s += v; s2 += v * v; }
    for (int off = 16; off > 0; off >>= 1) {
        s  += __shfl_down(s,  off, 32);
        s2 += __shfl_down(s2, off, 32);
    }
    __shared__ float sh[2][8];
    __shared__ float stats[2];
    int wave = threadIdx.x >> 5, lane = threadIdx.x & 31;
    if (lane == 0) { sh[0][wave] = s; sh[1][wave] = s2; }
    __syncthreads();
    if (threadIdx.x == 0) {
        float ts = 0.f, ts2 = 0.f;
        for (int i = 0; i < 8; ++i) { ts += sh[0][i]; ts2 += sh[1][i]; }
        float mu  = ts / (float)DM;
        float var = ts2 / (float)DM - mu * mu;
        stats[0] = mu; stats[1] = rsqrtf(var + 1e-5f);
    }
    __syncthreads();
    float mu = stats[0], rstd = stats[1];
    __bf16* o1 = hn     + (size_t)row * DM;
    __bf16* o2 = hn_rev + ((size_t)bb * LL + (size_t)(LL - 1 - l)) * DM;
    for (int i = threadIdx.x; i < DM; i += 256) {
        float v = (xr[i] - mu) * rstd * w[i] + bsh[i];
        __bf16 bv = (__bf16)v;
        o1[i] = bv; o2[i] = bv;
    }
}

// ---------------- generic f32 -> bf16 convert ----------------
__global__ void cvt_bf16_kernel(const float* __restrict__ s, __bf16* __restrict__ d, size_t n)
{
    size_t i = (size_t)blockIdx.x * blockDim.x + threadIdx.x;
    if (i < n) d[i] = (__bf16)s[i];
}

// ---------------- WMMA GEMM: C[M,N] = A[M,K](bf16) * W[N,K]^T(bf16) ----------------
// mode 0: store   mode 1: store softplus(v + bias[n])   mode 2: C += v
// Wave computes a 16(M) x 64(N) strip: one A fragment reused by 4 WMMAs.
// Block = 8 waves stacked along M -> 128(M) x 64(N) block tile.
// A 16x32 and B 32x16 fragments use contiguous k = 16*half + s (same order in both); f32 C 16x16.
__global__ __launch_bounds__(256) void gemm_bf16_kernel(
    const __bf16* __restrict__ A, const __bf16* __restrict__ W,
    float* __restrict__ C, int M, int N, int K,
    const float* __restrict__ bias, int mode)
{
    int lane = threadIdx.x & 31;
    int wave = threadIdx.x >> 5;
    int m0 = blockIdx.y * 128 + wave * 16;
    int n0 = blockIdx.x * 64;
    if (m0 >= M) return;                     // wave-uniform: EXEC stays all-ones

    int half = lane >> 4;
    const __bf16* ap = A + (size_t)(m0 + (lane & 15)) * K + 16 * half;
    const __bf16* wp = W + (size_t)(n0 + (lane & 15)) * K + 16 * half;
    size_t wstride = (size_t)16 * (size_t)K;   // 16 weight rows (one N tile)

    v8f acc[4];
#pragma unroll
    for (int nt = 0; nt < 4; ++nt) acc[nt] = (v8f){};

    for (int k0 = 0; k0 < K; k0 += 32) {
        union { v16bf v; v8bf h[2]; } au;
        au.h[0] = *(const v8bf*)(ap + k0);
        au.h[1] = *(const v8bf*)(ap + k0 + 8);
        __builtin_prefetch((const void*)(ap + k0 + 256), 0, 0);  // global_prefetch_b8
        v16bf b0 = *(const v16bf*)(wp + k0);
        v16bf b1 = *(const v16bf*)(wp + wstride + k0);
        v16bf b2 = *(const v16bf*)(wp + 2 * wstride + k0);
        v16bf b3 = *(const v16bf*)(wp + 3 * wstride + k0);
        acc[0] = __builtin_amdgcn_wmma_f32_16x16x32_bf16(false, au.v, false, b0,
                                                         (short)0, acc[0], false, false);
        acc[1] = __builtin_amdgcn_wmma_f32_16x16x32_bf16(false, au.v, false, b1,
                                                         (short)0, acc[1], false, false);
        acc[2] = __builtin_amdgcn_wmma_f32_16x16x32_bf16(false, au.v, false, b2,
                                                         (short)0, acc[2], false, false);
        acc[3] = __builtin_amdgcn_wmma_f32_16x16x32_bf16(false, au.v, false, b3,
                                                         (short)0, acc[3], false, false);
    }

    int crow = m0 + 8 * half;
    int ccol = n0 + (lane & 15);
#pragma unroll
    for (int nt = 0; nt < 4; ++nt) {
#pragma unroll
        for (int r = 0; r < 8; ++r) {
            float v = acc[nt][r];
            size_t idx = (size_t)(crow + r) * (size_t)N + (size_t)(ccol + nt * 16);
            if (mode == 1) {
                v += bias[ccol + nt * 16];
                v = (v > 20.f) ? v : log1pf(__expf(v));
                C[idx] = v;
            } else if (mode == 2) {
                C[idx] += v;
            } else {
                C[idx] = v;
            }
        }
    }
}

// ---------------- depthwise causal conv(4) + bias + SiLU ----------------
// xz layout: (B*L, 2048); channels [0,1024) are xm. Writes f32 + bf16 copies.
__global__ void conv_silu_kernel(const float* __restrict__ xz,
    const float* __restrict__ cw, const float* __restrict__ cb,
    float* __restrict__ xm2, __bf16* __restrict__ xm2b)
{
    size_t i = (size_t)blockIdx.x * 256 + threadIdx.x;
    if (i >= (size_t)MR * DI) return;
    int c = (int)(i % DI);
    size_t t = i / DI;
    int l  = (int)(t % LL);
    size_t bb = t / LL;
    float acc = cb[c];
#pragma unroll
    for (int j = 0; j < 4; ++j) {
        int ls = l - 3 + j;
        if (ls >= 0)
            acc += cw[c * 4 + j] * xz[((size_t)bb * LL + ls) * 2048 + c];
    }
    float sv = acc / (1.f + __expf(-acc));     // SiLU
    xm2[i]  = sv;
    xm2b[i] = (__bf16)sv;
}

// ---------------- slice dt (first 32 of xdbl's 64) -> bf16 ----------------
__global__ void slice_dt_kernel(const float* __restrict__ xdbl, __bf16* __restrict__ dtb)
{
    size_t i = (size_t)blockIdx.x * 256 + threadIdx.x;
    if (i >= (size_t)MR * DTR) return;
    size_t t = i / DTR;
    int r = (int)(i % DTR);
    dtb[i] = (__bf16)xdbl[t * 64 + r];
}

// ================= chunked associative scan (3 phases) =================
// Recurrence h = a*h + b with a = exp(delta*A) is associative per (channel,state).
// Carry layout: [(b*NC + chunk)*DS + s]*DI + d  -> lane-coalesced over d.

// Phase 1: per-chunk local scan from h=0; emit (prod a, h_end).
__global__ __launch_bounds__(256) void scan_pass1_kernel(const float* __restrict__ delta,
    const float* __restrict__ u, const float* __restrict__ xdbl,
    const float* __restrict__ A_log, float* __restrict__ aprod, float* __restrict__ hcarry)
{
    int idx = blockIdx.x * 256 + threadIdx.x;    // (b*NC + c)*DI + d
    if (idx >= BB * NC * DI) return;
    int d  = idx % DI;
    int c  = (idx / DI) % NC;
    int bb = idx / (DI * NC);
    float A[DS], ap[DS], h[DS];
#pragma unroll
    for (int s = 0; s < DS; ++s) {
        A[s] = -__expf(A_log[d * DS + s]);
        ap[s] = 1.f; h[s] = 0.f;
    }
    int l0 = c * CH;
    for (int l = l0; l < l0 + CH; ++l) {
        size_t t = (size_t)bb * LL + l;
        float dlt = delta[t * DI + d];
        float ut  = u[t * DI + d];
        const float* xr = xdbl + t * 64;
#pragma unroll
        for (int s = 0; s < DS; ++s) {
            float da = __expf(dlt * A[s]);
            ap[s] *= da;
            h[s] = da * h[s] + dlt * xr[DTR + s] * ut;
        }
    }
    size_t base = ((size_t)(bb * NC + c) * DS) * DI + d;
#pragma unroll
    for (int s = 0; s < DS; ++s) {
        aprod[base + (size_t)s * DI]  = ap[s];
        hcarry[base + (size_t)s * DI] = h[s];
    }
}

// Phase 2: sequential over NC chunk-carries (cheap); rewrite hcarry to h_init per chunk.
__global__ __launch_bounds__(256) void scan_pass2_kernel(const float* __restrict__ aprod,
    float* __restrict__ hcarry)
{
    int idx = blockIdx.x * 256 + threadIdx.x;    // b*DI + d
    if (idx >= BB * DI) return;
    int d  = idx % DI;
    int bb = idx / DI;
    float H[DS];
#pragma unroll
    for (int s = 0; s < DS; ++s) H[s] = 0.f;
    for (int c = 0; c < NC; ++c) {
        size_t base = ((size_t)(bb * NC + c) * DS) * DI + d;
#pragma unroll
        for (int s = 0; s < DS; ++s) {
            size_t k = base + (size_t)s * DI;
            float a    = aprod[k];
            float hend = hcarry[k];
            float newH = a * H[s] + hend;
            hcarry[k] = H[s];            // h_init for this chunk
            H[s] = newH;
        }
    }
}

// Phase 3: replay chunk with correct h_init; emit y.
__global__ __launch_bounds__(256) void scan_pass3_kernel(const float* __restrict__ delta,
    const float* __restrict__ u, const float* __restrict__ xdbl,
    const float* __restrict__ A_log, const float* __restrict__ hcarry,
    float* __restrict__ y)
{
    int idx = blockIdx.x * 256 + threadIdx.x;    // (b*NC + c)*DI + d
    if (idx >= BB * NC * DI) return;
    int d  = idx % DI;
    int c  = (idx / DI) % NC;
    int bb = idx / (DI * NC);
    float A[DS], h[DS];
    size_t base = ((size_t)(bb * NC + c) * DS) * DI + d;
#pragma unroll
    for (int s = 0; s < DS; ++s) {
        A[s] = -__expf(A_log[d * DS + s]);
        h[s] = hcarry[base + (size_t)s * DI];
    }
    int l0 = c * CH;
    for (int l = l0; l < l0 + CH; ++l) {
        size_t t = (size_t)bb * LL + l;
        float dlt = delta[t * DI + d];
        float ut  = u[t * DI + d];
        const float* xr = xdbl + t * 64;
        float yt = 0.f;
#pragma unroll
        for (int s = 0; s < DS; ++s) {
            float da = __expf(dlt * A[s]);
            h[s] = da * h[s] + dlt * xr[DTR + s] * ut;
            yt += h[s] * xr[DTR + DS + s];
        }
        y[t * DI + d] = yt;
    }
}

// ---------------- gating: yb = bf16((y + xm2*D) * silu(z)) ----------------
__global__ void gate_kernel(const float* __restrict__ y, const float* __restrict__ xm2,
    const float* __restrict__ Dp, const float* __restrict__ xz, __bf16* __restrict__ yb)
{
    size_t i = (size_t)blockIdx.x * 256 + threadIdx.x;
    if (i >= (size_t)MR * DI) return;
    size_t t = i / DI;
    int d = (int)(i % DI);
    float z  = xz[t * 2048 + DI + d];
    float sz = z / (1.f + __expf(-z));
    float v  = (y[i] + xm2[i] * Dp[d]) * sz;
    yb[i] = (__bf16)v;
}

// ---------------- final: out = gelu_exact(acc + x) ----------------
__global__ void final_kernel(const float* __restrict__ acc, const float* __restrict__ x,
                             float* __restrict__ out)
{
    size_t i = (size_t)blockIdx.x * 256 + threadIdx.x;
    if (i >= (size_t)MR * DM) return;
    float v = acc[i] + x[i];
    out[i] = 0.5f * v * (1.f + erff(v * 0.70710678118654752f));
}

// ---------------- host launch ----------------
extern "C" void kernel_launch(void* const* d_in, const int* in_sizes, int n_in,
                              void* d_out, int out_size, void* d_ws, size_t ws_size,
                              hipStream_t stream)
{
    const float* x    = (const float*)d_in[0];
    const float* ln_w = (const float*)d_in[19];
    const float* ln_b = (const float*)d_in[20];

    // carve workspace (256B aligned chunks)
    size_t off = 0;
    auto alloc = [&](size_t bytes) -> void* {
        void* p = (char*)d_ws + off;
        off += (bytes + 255) & ~(size_t)255;
        return p;
    };
    __bf16* hn_bf     = (__bf16*)alloc((size_t)MR * DM * 2);
    __bf16* hn_rev_bf = (__bf16*)alloc((size_t)MR * DM * 2);
    __bf16* inw_bf    = (__bf16*)alloc((size_t)2 * DI * DM * 2);
    __bf16* xw_bf     = (__bf16*)alloc((size_t)(DTR + 2 * DS) * DI * 2);
    __bf16* dtw_bf    = (__bf16*)alloc((size_t)DI * DTR * 2);
    __bf16* outw_bf   = (__bf16*)alloc((size_t)DM * DI * 2);
    float*  xz        = (float*)alloc((size_t)MR * 2 * DI * 4);
    float*  xm2_f     = (float*)alloc((size_t)MR * DI * 4);
    __bf16* xm2_bf    = (__bf16*)alloc((size_t)MR * DI * 2);
    float*  xdbl      = (float*)alloc((size_t)MR * 64 * 4);
    __bf16* dt_bf     = (__bf16*)alloc((size_t)MR * DTR * 2);
    float*  delta     = (float*)alloc((size_t)MR * DI * 4);
    float*  ybuf      = (float*)alloc((size_t)MR * DI * 4);
    __bf16* y_bf      = (__bf16*)alloc((size_t)MR * DI * 2);
    float*  accb      = (float*)alloc((size_t)MR * DM * 4);
    float*  aprod     = (float*)alloc((size_t)BB * NC * DI * DS * 4);
    float*  hcarry    = (float*)alloc((size_t)BB * NC * DI * DS * 4);
    (void)ws_size; (void)in_sizes; (void)n_in; (void)out_size;

    // layernorm + flipped copy
    ln_kernel<<<MR, 256, 0, stream>>>(x, ln_w, ln_b, hn_bf, hn_rev_bf);

    auto cvt = [&](const float* s, __bf16* d, size_t n) {
        cvt_bf16_kernel<<<(unsigned)((n + 255) / 256), 256, 0, stream>>>(s, d, n);
    };

    for (int dir = 0; dir < 2; ++dir) {
        const float* in_w   = (const float*)d_in[1 + dir * 9 + 0];
        const float* conv_w = (const float*)d_in[1 + dir * 9 + 1];
        const float* conv_b = (const float*)d_in[1 + dir * 9 + 2];
        const float* x_w    = (const float*)d_in[1 + dir * 9 + 3];
        const float* dt_w   = (const float*)d_in[1 + dir * 9 + 4];
        const float* dt_b   = (const float*)d_in[1 + dir * 9 + 5];
        const float* A_log  = (const float*)d_in[1 + dir * 9 + 6];
        const float* Dp     = (const float*)d_in[1 + dir * 9 + 7];
        const float* out_w  = (const float*)d_in[1 + dir * 9 + 8];

        cvt(in_w,  inw_bf,  (size_t)2 * DI * DM);
        cvt(x_w,   xw_bf,   (size_t)(DTR + 2 * DS) * DI);
        cvt(dt_w,  dtw_bf,  (size_t)DI * DTR);
        cvt(out_w, outw_bf, (size_t)DM * DI);

        const __bf16* actin = dir ? hn_rev_bf : hn_bf;

        // in-proj: (8192 x 512) * (2048 x 512)^T -> (8192 x 2048)
        {
            dim3 g(2 * DI / 64, MR / 128);
            gemm_bf16_kernel<<<g, 256, 0, stream>>>(actin, inw_bf, xz,
                                                    MR, 2 * DI, DM, nullptr, 0);
        }
        // conv + SiLU
        {
            size_t n = (size_t)MR * DI;
            conv_silu_kernel<<<(unsigned)((n + 255) / 256), 256, 0, stream>>>(
                xz, conv_w, conv_b, xm2_f, xm2_bf);
        }
        // x-proj: (8192 x 1024) * (64 x 1024)^T -> (8192 x 64)
        {
            dim3 g((DTR + 2 * DS) / 64, MR / 128);
            gemm_bf16_kernel<<<g, 256, 0, stream>>>(xm2_bf, xw_bf, xdbl,
                                                    MR, DTR + 2 * DS, DI, nullptr, 0);
        }
        // dt slice -> bf16
        {
            size_t n = (size_t)MR * DTR;
            slice_dt_kernel<<<(unsigned)((n + 255) / 256), 256, 0, stream>>>(xdbl, dt_bf);
        }
        // dt-proj + softplus(v + dt_b): (8192 x 32) * (1024 x 32)^T -> (8192 x 1024)
        {
            dim3 g(DI / 64, MR / 128);
            gemm_bf16_kernel<<<g, 256, 0, stream>>>(dt_bf, dtw_bf, delta,
                                                    MR, DI, DTR, dt_b, 1);
        }
        // chunked scan: phase1 (parallel), phase2 (tiny sequential), phase3 (parallel)
        scan_pass1_kernel<<<(BB * NC * DI + 255) / 256, 256, 0, stream>>>(
            delta, xm2_f, xdbl, A_log, aprod, hcarry);
        scan_pass2_kernel<<<(BB * DI + 255) / 256, 256, 0, stream>>>(aprod, hcarry);
        scan_pass3_kernel<<<(BB * NC * DI + 255) / 256, 256, 0, stream>>>(
            delta, xm2_f, xdbl, A_log, hcarry, ybuf);
        // gating
        {
            size_t n = (size_t)MR * DI;
            gate_kernel<<<(unsigned)((n + 255) / 256), 256, 0, stream>>>(
                ybuf, xm2_f, Dp, xz, y_bf);
        }
        // out-proj: (8192 x 1024) * (512 x 1024)^T -> acc (store for dir0, += for dir1)
        {
            dim3 g(DM / 64, MR / 128);
            gemm_bf16_kernel<<<g, 256, 0, stream>>>(y_bf, outw_bf, accb,
                                                    MR, DM, DI, nullptr, dir == 0 ? 0 : 2);
        }
    }

    // out = gelu(acc + x)
    {
        size_t n = (size_t)MR * DM;
        final_kernel<<<(unsigned)((n + 255) / 256), 256, 0, stream>>>(accb, x, (float*)d_out);
    }
}
